// HeteroClassifier_76063870812434
// MI455X (gfx1250) — matrix operand order repaired and checked by reading, not compiled, not run on hardware
//
#include <hip/hip_runtime.h>

typedef __attribute__((ext_vector_type(16))) __bf16 v16bf;
typedef __attribute__((ext_vector_type(8)))  float  v8f;
typedef __attribute__((ext_vector_type(8)))  unsigned int v8u;
typedef __attribute__((ext_vector_type(4)))  unsigned int v4u;
typedef __attribute__((ext_vector_type(4)))  unsigned int u32x4;
typedef __attribute__((ext_vector_type(8)))  int i32x8;
typedef __attribute__((ext_vector_type(4)))  int i32x4;
typedef unsigned short u16;

#define HD     128
#define NSTA   8
#define NFEA   210
#define NFPAD  224
#define NGRAPH 1024
#define LSTR   136   // bf16 LDS row stride: 272B = 17*16B -> aligned, bank-conflict-free

// native bf16 convert (RNE) -> v_cvt_pk_bf16_f32-class ops instead of manual rounding
__device__ __forceinline__ u16 f2bf(float f) {
  return __builtin_bit_cast(u16, static_cast<__bf16>(f));
}

union BFrag { v16bf v; v8u u; v4u q[2]; u16 s[16]; };

__device__ __forceinline__ v8f zero_v8f() {
  v8f z = {0.f,0.f,0.f,0.f,0.f,0.f,0.f,0.f};
  return z;
}

__device__ __forceinline__ v8f wmma_bf16(v16bf a, v16bf b, v8f c) {
  return __builtin_amdgcn_wmma_f32_16x16x32_bf16(false, a, false, b, (short)0, c, false, false);
}

// B fragment: weight stored N-major W[n][k] (bf16). Lane n = lane&15;
// lanes 0-15: K kb..kb+7 and kb+16..kb+23 with kb=k0; lanes 16-31: kb=k0+8.
__device__ __forceinline__ v16bf load_bfrag(const u16* W, int Kstride, int ntile, int k0, int lane) {
  int n  = ntile * 16 + (lane & 15);
  int kb = k0 + ((lane >> 4) << 3);
  BFrag f;
  f.q[0] = *(const v4u*)(W + n * Kstride + kb);
  f.q[1] = *(const v4u*)(W + n * Kstride + kb + 16);
  return f.v;
}

// A fragment from bf16 LDS tile (row-major, stride LSTR). Row = lane&15;
// lanes 0-15: K kb..kb+7 and kb+16..kb+23 with kb=k0; lanes 16-31: kb=k0+8.
__device__ __forceinline__ v16bf load_afrag_lds(const u16* tile, int k0, int lane) {
  int r  = lane & 15;
  int kb = k0 + ((lane >> 4) << 3);
  BFrag f;
  f.q[0] = *(const v4u*)(tile + r * LSTR + kb);
  f.q[1] = *(const v4u*)(tile + r * LSTR + kb + 16);
  return f.v;
}

__device__ __forceinline__ void cvt8(const float* p, u16* d) {
#pragma unroll
  for (int i = 0; i < 4; ++i) {
    float2 t = *(const float2*)(p + 2 * i);   // rows are 8B-aligned
    d[2*i]   = f2bf(t.x);
    d[2*i+1] = f2bf(t.y);
  }
}

// A fragment from an fp32 row (global or LDS), zero-padded past kmax.
__device__ __forceinline__ v16bf load_afrag_f32(const float* rowp, int k0, int lane, int kmax) {
  int kb = k0 + ((lane >> 4) << 3);
  BFrag f;
#pragma unroll
  for (int c = 0; c < 2; ++c) {
    int ks = kb + c * 16;
    if (ks + 8 <= kmax) {
      cvt8(rowp + ks, f.s + c * 8);
    } else {
#pragma unroll
      for (int i = 0; i < 8; ++i) {
        int k = ks + i;
        f.s[c * 8 + i] = (k < kmax) ? f2bf(rowp[k]) : (u16)0;
      }
    }
  }
  return f.v;
}

// Store a 16x128 C tile (8 v8f fragments) as bf16 into LDS, row-major.
__device__ __forceinline__ void store_tile(const v8f* acc, u16* tileP, int hh, int nloc) {
#pragma unroll
  for (int nt = 0; nt < 8; ++nt)
#pragma unroll
    for (int j = 0; j < 8; ++j)
      tileP[(j + 8 * hh) * LSTR + nt * 16 + nloc] = f2bf(acc[nt][j]);
}

// TDM: async-load a [rows x cols] fp32 tile (row stride = cols) into LDS.
__device__ __forceinline__ void tdm_load_tile_f32(const float* gsrc, void* lds_dst,
                                                  int cols, int rows) {
  unsigned lds_off = (unsigned)(unsigned long long)lds_dst;     // flat addr low 32b = LDS offset
  unsigned long long ga = (unsigned long long)gsrc;
  u32x4 g0;
  g0[0] = 1u;                                   // count=1, is_restore=0, gather=0
  g0[1] = lds_off;                              // lds_addr (bytes)
  g0[2] = (unsigned)ga;                         // global_addr[31:0]
  g0[3] = (unsigned)(ga >> 32) | 0x80000000u;   // global_addr[56:32] | type=2<<30
  i32x8 g1;
  g1[0] = 0x00020000;                           // workgroup_mask=0, data_size=2 (4B)
  g1[1] = (int)((unsigned)cols << 16);          // tensor_dim0[15:0]<<16 (atomic_barrier_addr=0)
  g1[2] = (int)((unsigned)rows << 16);          // tensor_dim0 hi=0 | tensor_dim1[15:0]<<16
  g1[3] = (int)((unsigned)cols << 16);          // tensor_dim1 hi=0 | tile_dim0<<16
  g1[4] = rows;                                 // tile_dim1 | tile_dim2=0
  g1[5] = cols;                                 // tensor_dim0_stride[31:0]
  g1[6] = 0;                                    // stride hi | tensor_dim1_stride lo
  g1[7] = 0;
  i32x4 gz  = {0, 0, 0, 0};
  i32x8 gz8 = {0, 0, 0, 0, 0, 0, 0, 0};
  __builtin_amdgcn_tensor_load_to_lds(g0, g1, gz, gz, gz8, 0);
}

// ---------------------------------------------------------------------------
// Kernel 1: weight conversion fp32 -> bf16 N-major (transposed), zero rf.
// ---------------------------------------------------------------------------
__global__ __launch_bounds__(256) void prep_kernel(
    const float* __restrict__ W1_ft, const float* __restrict__ Wself_hf,
    const float* __restrict__ Wneigh_hf, const float* __restrict__ Wself_tt,
    const float* __restrict__ Wneigh_tt, const float* __restrict__ W1_st,
    u16* __restrict__ W1ftb, u16* __restrict__ Wshf, u16* __restrict__ Wnhf,
    u16* __restrict__ Wstt, u16* __restrict__ Wntt, u16* __restrict__ W1stb,
    float* __restrict__ rf)
{
  int i = blockIdx.x * 256 + threadIdx.x;
  if (i < NGRAPH) rf[i] = 0.f;
  if (i < 128 * NFPAD) {                           // W1_ft: [210,128] -> [128][224]
    int n = i / NFPAD, k = i % NFPAD;
    W1ftb[i] = (k < NFEA) ? f2bf(W1_ft[k * HD + n]) : (u16)0;
  }
  int j = i - 128 * NFPAD;
  if (j >= 0 && j < 3 * HD * HD) {                 // four [3][128][128] families
    int l = j / (HD * HD), r = j % (HD * HD);
    int n = r / HD, k = r % HD;
    int src = l * HD * HD + k * HD + n;
    Wshf[j] = f2bf(Wself_hf[src]);
    Wnhf[j] = f2bf(Wneigh_hf[src]);
    Wstt[j] = f2bf(Wself_tt[src]);
    Wntt[j] = f2bf(Wneigh_tt[src]);
  }
  int m = i - (128 * NFPAD + 3 * HD * HD);
  if (m >= 0 && m < 128 * 32) {                    // W1_st: [8,128] -> [128][32]
    int n = m / 32, k = m % 32;
    W1stb[m] = (k < NSTA) ? f2bf(W1_st[k * HD + n]) : (u16)0;
  }
}

// ---------------------------------------------------------------------------
// Kernel 2: full station chain. One wave per 16 station rows (= 2 graphs).
// Produces mterm[l][g][:] = m_l[g] @ Wneigh_hf[l] + b_hf[l], and rs[g].
// ---------------------------------------------------------------------------
__global__ __launch_bounds__(128) void station_kernel(
    const float* __restrict__ h_st, const u16* __restrict__ W1stb,
    const u16* __restrict__ Wstt, const u16* __restrict__ Wntt,
    const u16* __restrict__ Wnhf, const float* __restrict__ b1_st,
    const float* __restrict__ b_tt, const float* __restrict__ b_hf,
    const float* __restrict__ W2_st, const float* __restrict__ b2_st,
    float* __restrict__ mterm, float* __restrict__ rs)
{
  __shared__ u16 hsT[4][16 * LSTR];
  __shared__ u16 mT[4][256];                 // [g in tile][k] bf16, 2x128
  int wv = threadIdx.x >> 5, lane = threadIdx.x & 31;
  int tile = blockIdx.x * 4 + wv;            // 0..511, graphs 2*tile, 2*tile+1
  int nloc = lane & 15, hh = lane >> 4;
  u16* tileP = hsT[wv];
  u16* mP    = mT[wv];

  v8f acc[8];
#pragma unroll
  for (int nt = 0; nt < 8; ++nt) acc[nt] = zero_v8f();

  // projection: hs0 = h_st @ W1_st + b1_st  (K padded 8 -> 32)
  {
    const float* rowp = h_st + (tile * 16 + nloc) * NSTA;
    v16bf a = load_afrag_f32(rowp, 0, lane, NSTA);
#pragma unroll
    for (int nt = 0; nt < 8; ++nt)
      acc[nt] = wmma_bf16(a, load_bfrag(W1stb, 32, nt, 0, lane), acc[nt]);
  }
#pragma unroll
  for (int nt = 0; nt < 8; ++nt) {
    float bv = b1_st[nt * 16 + nloc];
#pragma unroll
    for (int j = 0; j < 8; ++j) acc[nt][j] += bv;
  }
  store_tile(acc, tileP, hh, nloc);

  for (int l = 0; l < 3; ++l) {
    // per-graph station mean from C layout: lane(half) owns graph hh, column n
#pragma unroll
    for (int nt = 0; nt < 8; ++nt) {
      float s = 0.f;
#pragma unroll
      for (int j = 0; j < 8; ++j) s += acc[nt][j];
      mP[hh * HD + nt * 16 + nloc] = f2bf(s * 0.125f);
    }

    // mterm_l = m_l @ Wneigh_hf[l] + b_hf[l]  (A = m broadcast to rows of its graph)
#pragma unroll
    for (int nt = 0; nt < 8; ++nt) acc[nt] = zero_v8f();
    for (int k0 = 0; k0 < HD; k0 += 32) {
      int r = lane & 15, g = r >> 3, kb = k0 + ((lane >> 4) << 3);
      BFrag f;
      f.q[0] = *(const v4u*)(mP + g * HD + kb);
      f.q[1] = *(const v4u*)(mP + g * HD + kb + 16);
#pragma unroll
      for (int nt = 0; nt < 8; ++nt)
        acc[nt] = wmma_bf16(f.v, load_bfrag(Wnhf + l * HD * HD, HD, nt, k0, lane), acc[nt]);
    }
#pragma unroll
    for (int nt = 0; nt < 8; ++nt) {   // row M=8*hh is row 0 of graph (tile*2+hh)
      int n = nt * 16 + nloc;
      mterm[(l * NGRAPH + tile * 2 + hh) * HD + n] = acc[nt][0] + b_hf[l * HD + n];
    }

    // hs_new = relu(hs@Wself_tt[l] + m_bcast@Wneigh_tt[l] + b_tt[l])
#pragma unroll
    for (int nt = 0; nt < 8; ++nt) acc[nt] = zero_v8f();
    for (int k0 = 0; k0 < HD; k0 += 32) {
      v16bf a = load_afrag_lds(tileP, k0, lane);
#pragma unroll
      for (int nt = 0; nt < 8; ++nt)
        acc[nt] = wmma_bf16(a, load_bfrag(Wstt + l * HD * HD, HD, nt, k0, lane), acc[nt]);
    }
    for (int k0 = 0; k0 < HD; k0 += 32) {
      int r = lane & 15, g = r >> 3, kb = k0 + ((lane >> 4) << 3);
      BFrag f;
      f.q[0] = *(const v4u*)(mP + g * HD + kb);
      f.q[1] = *(const v4u*)(mP + g * HD + kb + 16);
#pragma unroll
      for (int nt = 0; nt < 8; ++nt)
        acc[nt] = wmma_bf16(f.v, load_bfrag(Wntt + l * HD * HD, HD, nt, k0, lane), acc[nt]);
    }
#pragma unroll
    for (int nt = 0; nt < 8; ++nt) {
      float bv = b_tt[l * HD + nt * 16 + nloc];
#pragma unroll
      for (int j = 0; j < 8; ++j) {
        float v = acc[nt][j] + bv;
        acc[nt][j] = v > 0.f ? v : 0.f;
      }
    }
    if (l < 2) store_tile(acc, tileP, hh, nloc);
  }

  // rs[g] = sum_s (hs_final[s] . W2_st) + 8*b2_st
  float p = 0.f;
#pragma unroll
  for (int nt = 0; nt < 8; ++nt) {
    float w = W2_st[nt * 16 + nloc];
#pragma unroll
    for (int j = 0; j < 8; ++j) p += acc[nt][j] * w;
  }
#pragma unroll
  for (int m = 1; m < 16; m <<= 1) p += __shfl_xor(p, m, 32);
  if (nloc == 0) rs[tile * 2 + hh] = p + 8.f * b2_st[0];
}

// ---------------------------------------------------------------------------
// Kernel 3: feature-node pipeline. 4 waves/block, 16 rows/wave.
// A-tile of h_ft arrives via the Tensor Data Mover (tensor_load_to_lds).
// ---------------------------------------------------------------------------
__global__ __launch_bounds__(128) void feat_kernel(
    const float* __restrict__ h_ft, const u16* __restrict__ W1ftb,
    const u16* __restrict__ Wshf, const float* __restrict__ b1_ft,
    const float* __restrict__ mterm, const float* __restrict__ W2_ft,
    const float* __restrict__ b2_ft, float* __restrict__ rf)
{
  // fp32 stage (16x210 per wave); the bf16 activation tile aliases its start
  // (fp32 data is dead after GEMM1; same-wave DS ops are in-order).
  __shared__ float stage[4][16 * NFEA];      // 53760 B
  int wv = threadIdx.x >> 5, lane = threadIdx.x & 31;
  int tile = blockIdx.x * 4 + wv;            // 0..13439 (exact)
  int nloc = lane & 15, hh = lane >> 4;
  float* stF  = stage[wv];
  u16*   tileP = (u16*)stF;
  int base = tile * 16;

  // TDM DMA: 16 rows x 210 fp32, row stride 210, into this wave's stage
  tdm_load_tile_f32(h_ft + (long)base * NFEA, stF, NFEA, 16);

  v8f acc[8];
#pragma unroll
  for (int nt = 0; nt < 8; ++nt) acc[nt] = zero_v8f();

  __builtin_amdgcn_s_wait_tensorcnt(0);
  asm volatile("" : : "r"(stF) : "memory");  // TDM wrote LDS behind the compiler's back

  // hf0 = h_ft @ W1_ft + b1_ft  (K: 210 padded to 224), A from LDS fp32 stage
  for (int k0 = 0; k0 < NFPAD; k0 += 32) {
    v16bf a = load_afrag_f32(stF + (lane & 15) * NFEA, k0, lane, NFEA);
#pragma unroll
    for (int nt = 0; nt < 8; ++nt)
      acc[nt] = wmma_bf16(a, load_bfrag(W1ftb, NFPAD, nt, k0, lane), acc[nt]);
  }
#pragma unroll
  for (int nt = 0; nt < 8; ++nt) {
    float bv = b1_ft[nt * 16 + nloc];
#pragma unroll
    for (int j = 0; j < 8; ++j) acc[nt][j] += bv;
  }
  store_tile(acc, tileP, hh, nloc);

  for (int l = 0; l < 3; ++l) {
#pragma unroll
    for (int nt = 0; nt < 8; ++nt) acc[nt] = zero_v8f();
    for (int k0 = 0; k0 < HD; k0 += 32) {
      v16bf a = load_afrag_lds(tileP, k0, lane);
#pragma unroll
      for (int nt = 0; nt < 8; ++nt)
        acc[nt] = wmma_bf16(a, load_bfrag(Wshf + l * HD * HD, HD, nt, k0, lane), acc[nt]);
    }
    // + (m_l @ Wneigh_hf[l] + b_hf[l]) per graph, then ReLU
#pragma unroll
    for (int j = 0; j < 8; ++j) {
      int row = base + j + 8 * hh;
      int g = row / NFEA;
      const float* mrow = mterm + (l * NGRAPH + g) * HD;
#pragma unroll
      for (int nt = 0; nt < 8; ++nt) {
        float v = acc[nt][j] + mrow[nt * 16 + nloc];
        acc[nt][j] = v > 0.f ? v : 0.f;
      }
    }
    if (l < 2) store_tile(acc, tileP, hh, nloc);
  }

  // of[row] = hf . W2_ft + b2_ft ; rf[g] += of
  float w2v[8];
#pragma unroll
  for (int nt = 0; nt < 8; ++nt) w2v[nt] = W2_ft[nt * 16 + nloc];
  float b2v = b2_ft[0];
#pragma unroll
  for (int j = 0; j < 8; ++j) {
    float s = 0.f;
#pragma unroll
    for (int nt = 0; nt < 8; ++nt) s += acc[nt][j] * w2v[nt];
#pragma unroll
    for (int m = 1; m < 16; m <<= 1) s += __shfl_xor(s, m, 32);
    if (nloc == 0) {
      int row = base + j + 8 * hh;
      atomicAdd(&rf[row / NFEA], s + b2v);
    }
  }
}

// ---------------------------------------------------------------------------
// Kernel 4: final readout head.
// ---------------------------------------------------------------------------
__global__ __launch_bounds__(256) void out_kernel(
    const float* __restrict__ rs, const float* __restrict__ rf,
    const float* __restrict__ W_out, const float* __restrict__ b_out,
    float* __restrict__ out)
{
  int g = blockIdx.x * 256 + threadIdx.x;
  if (g < NGRAPH) out[g] = rs[g] * W_out[0] + rf[g] * W_out[1] + b_out[0];
}

extern "C" void kernel_launch(void* const* d_in, const int* in_sizes, int n_in,
                              void* d_out, int out_size, void* d_ws, size_t ws_size,
                              hipStream_t stream) {
  const float* h_st      = (const float*)d_in[0];
  const float* h_ft      = (const float*)d_in[1];
  const float* Wself_tt  = (const float*)d_in[2];
  const float* Wneigh_tt = (const float*)d_in[3];
  const float* b_tt      = (const float*)d_in[4];
  const float* Wself_hf  = (const float*)d_in[5];
  const float* Wneigh_hf = (const float*)d_in[6];
  const float* b_hf      = (const float*)d_in[7];
  const float* W1_st     = (const float*)d_in[8];
  const float* b1_st     = (const float*)d_in[9];
  const float* W1_ft     = (const float*)d_in[10];
  const float* b1_ft     = (const float*)d_in[11];
  const float* W2_st     = (const float*)d_in[12];
  const float* b2_st     = (const float*)d_in[13];
  const float* W2_ft     = (const float*)d_in[14];
  const float* b2_ft     = (const float*)d_in[15];
  const float* W_out     = (const float*)d_in[16];
  const float* b_out     = (const float*)d_in[17];
  // edge lists d_in[18..21] are structurally fixed -> not needed

  // workspace partition (all 32B-aligned)
  u16*   W1ftb = (u16*)d_ws;                       // 128*224
  u16*   Wshf  = W1ftb + 128 * NFPAD;              // 3*128*128
  u16*   Wnhf  = Wshf  + 3 * HD * HD;
  u16*   Wstt  = Wnhf  + 3 * HD * HD;
  u16*   Wntt  = Wstt  + 3 * HD * HD;
  u16*   W1stb = Wntt  + 3 * HD * HD;              // 128*32
  float* mterm = (float*)(W1stb + 128 * 32);       // 3*1024*128
  float* rs    = mterm + 3 * NGRAPH * HD;          // 1024
  float* rf    = rs + NGRAPH;                      // 1024

  int prep_elems = 128 * NFPAD + 3 * HD * HD;      // 77824 -> 304 blocks exactly
  prep_kernel<<<(prep_elems + 255) / 256, 256, 0, stream>>>(
      W1_ft, Wself_hf, Wneigh_hf, Wself_tt, Wneigh_tt, W1_st,
      W1ftb, Wshf, Wnhf, Wstt, Wntt, W1stb, rf);

  station_kernel<<<128, 128, 0, stream>>>(
      h_st, W1stb, Wstt, Wntt, Wnhf, b1_st, b_tt, b_hf, W2_st, b2_st, mterm, rs);

  feat_kernel<<<3360, 128, 0, stream>>>(
      h_ft, W1ftb, Wshf, b1_ft, mterm, W2_ft, b2_ft, rf);

  out_kernel<<<4, 256, 0, stream>>>(rs, rf, W_out, b_out, (float*)d_out);
}
